// BiLSTM_CRF_7310034338575
// MI455X (gfx1250) — matrix operand order repaired and checked by reading, not compiled, not run on hardware
//
#include <hip/hip_runtime.h>
#include <hip/hip_bf16.h>
#include <stdint.h>

// ---------------- problem constants ----------------
#define T_LEN 4096
#define E_DIM 256
#define H_DIM 512
#define G4    2048      // 4*H
#define K_TAG 16
#define NBLK  16        // workgroups per direction in recurrence
#define SLICE 32        // h elements per workgroup (H / NBLK)
#define ROWS  128       // gate rows per workgroup (4*SLICE)
#define RDY_STRIDE 16   // ints per ready flag (64B line padding)
#define START_TAG 14
#define END_TAG   15
#define NEG_INF  (-10000.0f)

typedef __bf16 v8bf  __attribute__((ext_vector_type(8)));
typedef __bf16 v16bf __attribute__((ext_vector_type(16)));
typedef float  v8f   __attribute__((ext_vector_type(8)));
typedef unsigned int v4u __attribute__((ext_vector_type(4)));
typedef int    v4i   __attribute__((ext_vector_type(4)));
typedef int    v8i   __attribute__((ext_vector_type(8)));

static __device__ __forceinline__ float sigmoidf(float x) {
  return 1.0f / (1.0f + __expf(-x));
}

static __device__ __forceinline__ v16bf cat8(v8bf a, v8bf b) {
  return __builtin_shufflevector(a, b, 0,1,2,3,4,5,6,7,8,9,10,11,12,13,14,15);
}

static __device__ __forceinline__ v8f wmma_bf16(v16bf a, v16bf b, v8f c) {
  // D = A(16x32 bf16) x B(32x16 bf16) + C(16x16 f32)
  return __builtin_amdgcn_wmma_f32_16x16x32_bf16(false, a, false, b, (short)0, c,
                                                 false, false);
}

// ---------------- phase 1: conversions / gather ----------------
__global__ void k_cvt_rm(__bf16* __restrict__ dst, const float* __restrict__ src, int n) {
  int i = blockIdx.x * blockDim.x + threadIdx.x;
  if (i < n) dst[i] = (__bf16)src[i];
}

// Whh [4H=2048, 512] f32 -> bf16 permuted as [wb=16][gate=4][j=32][k=512] so each
// recurrence workgroup's slice is one contiguous 128 KB block (single TDM tile).
__global__ void k_cvt_whh(__bf16* __restrict__ dst, const float* __restrict__ src) {
  int i = blockIdx.x * blockDim.x + threadIdx.x;
  if (i < G4 * H_DIM) {
    int k = i & 511;
    int r = i >> 9;          // dest row index: ((wb*4+q)*32 + j)
    int j = r & 31;
    int q = (r >> 5) & 3;
    int wb = r >> 7;
    int srow = q * H_DIM + wb * SLICE + j;
    dst[i] = (__bf16)src[(size_t)srow * H_DIM + k];
  }
}

__global__ void k_gather(__bf16* __restrict__ dst, const float* __restrict__ embed,
                         const int* __restrict__ x) {
  int i = blockIdx.x * blockDim.x + threadIdx.x;   // T*E
  if (i < T_LEN * E_DIM) {
    int t = i >> 8;
    int e = i & 255;
    dst[i] = (__bf16)embed[(size_t)x[t] * E_DIM + e];
  }
}

// ---------------- phase 2: input-projection GEMM (WMMA bf16) ----------------
// G[dir] = emb[T,256] @ Wih^T[256,2048] + b   (f32 out, parked in L2: 2x32MB)
__global__ __launch_bounds__(256) void k_gemm_in(const __bf16* __restrict__ embbf,
                                                 const __bf16* __restrict__ wih_f,
                                                 const __bf16* __restrict__ wih_b,
                                                 const float* __restrict__ bias_f,
                                                 const float* __restrict__ bias_b,
                                                 float* __restrict__ Gf,
                                                 float* __restrict__ Gb) {
  const int lane = threadIdx.x & 31;
  const int wv = threadIdx.x >> 5;
  const int gw = blockIdx.x * 8 + wv;          // 0..65535 waves, 1 tile each
  const int dir = gw >> 15;
  const int r = gw & 32767;
  const int mt = r >> 7;                       // 0..255 (T/16)
  const int nt = r & 127;                      // 0..127 (2048/16)
  const int lo = lane & 15, hi = lane >> 4;
  const __bf16* wih = dir ? wih_b : wih_f;
  const float* bias = dir ? bias_b : bias_f;
  float* G = dir ? Gb : Gf;

  const __bf16* arow = embbf + (size_t)(mt * 16 + lo) * E_DIM;
  const __bf16* brow = wih + (size_t)(nt * 16 + lo) * E_DIM;
  v8f acc = {};
#pragma unroll
  for (int kt = 0; kt < E_DIM / 32; ++kt) {
    v8bf a0 = *(const v8bf*)(arow + kt * 32 + hi * 16);
    v8bf a1 = *(const v8bf*)(arow + kt * 32 + hi * 16 + 8);
    v16bf b = *(const v16bf*)(brow + kt * 32 + hi * 16);
    acc = wmma_bf16(cat8(a0, a1), b, acc);
  }
  const int n = nt * 16 + lo;
  const float bv = bias[n];
#pragma unroll
  for (int vv = 0; vv < 8; ++vv) {
    const int m = mt * 16 + vv + 8 * hi;       // C/D layout: M = v + 8*(lane/16)
    G[(size_t)m * G4 + n] = acc[vv] + bv;
  }
}

// ---------------- phase 3: init for recurrence ----------------
__global__ void k_init(int* __restrict__ ready, __bf16* __restrict__ hbf_f,
                       __bf16* __restrict__ hbf_b, const float* __restrict__ h0) {
  int i = blockIdx.x * blockDim.x + threadIdx.x;
  if (i < 2 * (T_LEN + 1) * RDY_STRIDE) {
    int within = i % ((T_LEN + 1) * RDY_STRIDE);
    ready[i] = (within == 0) ? NBLK : 0;       // step-0 flags pre-completed
  }
  if (i < 2 * H_DIM) {
    int dir = i >> 9, j = i & 511;
    (dir ? hbf_b : hbf_f)[j] = (__bf16)h0[i];
  }
}

// ---------------- phase 4: sequential BiLSTM recurrence ----------------
// 32 persistent workgroups (16 per direction). Each holds a contiguous 128x512
// bf16 Whh slice in LDS (TDM-preloaded), computes its 128 gate pre-activations
// with WMMA matvecs (h broadcast across M), owns 32 cell states, and exchanges
// the 512-wide h vector (bf16) through L2 with a release/acquire flag per step.
__global__ __launch_bounds__(256) void k_rec(const __bf16* __restrict__ whh_f,
                                             const __bf16* __restrict__ whh_b,
                                             const float* __restrict__ Gf,
                                             const float* __restrict__ Gb,
                                             const float* __restrict__ c0,
                                             __bf16* __restrict__ hbf_f,
                                             __bf16* __restrict__ hbf_b,
                                             int* __restrict__ ready) {
  extern __shared__ char smem[];
  __bf16* whh_lds = (__bf16*)smem;                 // 128*512*2 = 131072 B
  __bf16* hbf     = (__bf16*)(smem + 131072);      // 512*2 B
  float*  ysm     = (float*)(smem + 132096);       // 128*4 B

  const int tid = threadIdx.x;
  const int dir = blockIdx.x >> 4;
  const int wb  = blockIdx.x & 15;
  const int base = wb * SLICE;
  const __bf16* wsrc = (dir ? whh_b : whh_f) + (size_t)wb * ROWS * H_DIM;
  const float* Gd = dir ? Gb : Gf;

  // ---- one-time Whh slice preload into LDS (128 KB) ----
#if defined(__gfx1250__) && __has_builtin(__builtin_amdgcn_tensor_load_to_lds)
  if (tid == 0) {
    unsigned long long ga = (unsigned long long)(const void*)wsrc;
    unsigned lds = (unsigned)(unsigned long long)(void*)whh_lds; // low 32b = LDS offset
    v4u g0 = { 1u,                                   // count=1 valid descriptor
               lds,
               (unsigned)(ga & 0xffffffffull),
               (unsigned)(((ga >> 32) & 0x1ffffffull) | (2u << 30)) }; // type=2
    v8i g1 = { (int)(1u << 16),    // data_size = 1 (2-byte elems)
               (int)(512u << 16),  // tensor_dim0 = 512
               (int)(128u << 16),  // tensor_dim1 = 128
               (int)(512u << 16),  // tile_dim0   = 512
               128,                // tile_dim1   = 128
               512,                // tensor_dim0_stride = 512 (contiguous)
               0, 0 };
    v4i gz4 = { 0, 0, 0, 0 };
#if __clang_major__ >= 23
    v8i gz8 = { 0, 0, 0, 0, 0, 0, 0, 0 };
    __builtin_amdgcn_tensor_load_to_lds(g0, g1, gz4, gz4, gz8, 0);
#else
    __builtin_amdgcn_tensor_load_to_lds(g0, g1, gz4, gz4, 0);
#endif
#if __has_builtin(__builtin_amdgcn_s_wait_tensorcnt)
    __builtin_amdgcn_s_wait_tensorcnt(0);
#else
    asm volatile("s_wait_tensorcnt 0x0" ::: "memory");
#endif
  }
#else
  {
    const uint4* s = (const uint4*)wsrc;
    uint4* d = (uint4*)whh_lds;
    for (int i = tid; i < (ROWS * H_DIM * 2) / 16; i += 256) d[i] = s[i];
  }
#endif
  __syncthreads();

  const int lane = tid & 31;
  const int wv = tid >> 5;
  const int lo = lane & 15;
  const int hi = lane >> 4;
  const int hj = base + (tid & 31);            // valid for tid < SLICE

  float c_reg = (tid < SLICE) ? c0[dir * H_DIM + base + tid] : 0.0f;

  int* rdy = ready + (size_t)dir * (T_LEN + 1) * RDY_STRIDE;
  const __bf16* hrd = dir ? hbf_b : hbf_f;
  __bf16* hwr = dir ? hbf_b : hbf_f;

  for (int s = 0; s < T_LEN; ++s) {
    // ---- G for this step is h-independent: issue loads BEFORE the spin so the
    //      L2 fetch overlaps the wait; prefetch next step's lines too.
    const int t_idx = dir ? (T_LEN - 1 - s) : s;
    const float* grow = Gd + (size_t)t_idx * G4;
    float gi = 0.f, gf = 0.f, gg = 0.f, go = 0.f;
    if (tid < SLICE) {
      gi = grow[0 * H_DIM + hj];
      gf = grow[1 * H_DIM + hj];
      gg = grow[2 * H_DIM + hj];
      go = grow[3 * H_DIM + hj];
      if (s + 1 < T_LEN) {
        const float* gn = Gd + (size_t)(dir ? (t_idx - 1) : (t_idx + 1)) * G4;
        __builtin_prefetch(gn + 0 * H_DIM + hj, 0, 3);
        __builtin_prefetch(gn + 1 * H_DIM + hj, 0, 3);
        __builtin_prefetch(gn + 2 * H_DIM + hj, 0, 3);
        __builtin_prefetch(gn + 3 * H_DIM + hj, 0, 3);
      }
    }

    // ---- wait for full h[s] (all 16 slices published) ----
    if (tid == 0) {
      while (__hip_atomic_load(rdy + (size_t)s * RDY_STRIDE, __ATOMIC_ACQUIRE,
                               __HIP_MEMORY_SCOPE_AGENT) < NBLK)
        __builtin_amdgcn_s_sleep(1);
    }
    __syncthreads();

    // ---- stage h[s] (bf16, 1 KB) into LDS: 256 threads x 1 dword ----
    ((unsigned*)hbf)[tid] = ((const unsigned*)(hrd + (size_t)s * H_DIM))[tid];
    __syncthreads();

    // ---- y = Whh_slice @ h via WMMA: wave wv owns gate rows wv*16..+15 ----
    v8f acc = {};
    const __bf16* brow = whh_lds + (size_t)(wv * 16 + lo) * H_DIM;
#pragma unroll
    for (int kt = 0; kt < H_DIM / 32; ++kt) {
      v8bf a0 = *(const v8bf*)(hbf + kt * 32 + hi * 16);   // h chunk (A rows replicated)
      v8bf a1 = *(const v8bf*)(hbf + kt * 32 + hi * 16 + 8);
      v16bf b = *(const v16bf*)(brow + kt * 32 + hi * 16);
      acc = wmma_bf16(cat8(a0, a1), b, acc);
    }
    if (hi == 0) ysm[wv * 16 + lo] = acc[0];  // every D row equals y; take M=0
    __syncthreads();

    // ---- gate nonlinearities + cell update on wave 0 (lanes 0..31) ----
    if (tid < SLICE) {
      float yi = ysm[0 * SLICE + tid] + gi;
      float yf = ysm[1 * SLICE + tid] + gf;
      float yg = ysm[2 * SLICE + tid] + gg;
      float yo = ysm[3 * SLICE + tid] + go;
      c_reg = sigmoidf(yf) * c_reg + sigmoidf(yi) * tanhf(yg);
      hwr[(size_t)(s + 1) * H_DIM + hj] = (__bf16)(sigmoidf(yo) * tanhf(c_reg));
    }
    // writers and signaler are the same wave: fence orders the 64B h store
    // before the release atomic; no extra __syncthreads needed.
    if (tid == 0) {
      __threadfence();
      __hip_atomic_fetch_add(rdy + (size_t)(s + 1) * RDY_STRIDE, 1,
                             __ATOMIC_RELEASE, __HIP_MEMORY_SCOPE_AGENT);
    }
  }
}

// ---------------- phase 5: FC GEMM [T,1024]x[1024,16] (WMMA) ----------------
// A-fragments gathered directly from the bf16 h streams (no concat pass).
__global__ __launch_bounds__(256) void k_gemm_fc(const __bf16* __restrict__ hbf_f,
                                                 const __bf16* __restrict__ hbf_b,
                                                 const __bf16* __restrict__ fcw,
                                                 const float* __restrict__ fcb,
                                                 float* __restrict__ feats) {
  const int lane = threadIdx.x & 31;
  const int wv = threadIdx.x >> 5;
  const int mt = blockIdx.x * 8 + wv;              // 0..255
  const int lo = lane & 15, hi = lane >> 4;
  const int m = mt * 16 + lo;                      // timestep this lane's A-row covers
  const __bf16* af = hbf_f + (size_t)(m + 1) * H_DIM;        // hf[t] = hseq_f[t+1]
  const __bf16* ab = hbf_b + (size_t)(T_LEN - m) * H_DIM;    // hb[t] = hseq_b[T-t]
  const __bf16* brow = fcw + (size_t)lo * (2 * H_DIM);
  v8f acc = {};
  for (int kt = 0; kt < (2 * H_DIM) / 32; ++kt) {
    const __bf16* ar = (kt < 16) ? (af + kt * 32) : (ab + (kt - 16) * 32);
    v8bf a0 = *(const v8bf*)(ar + hi * 16);
    v8bf a1 = *(const v8bf*)(ar + hi * 16 + 8);
    v16bf b = *(const v16bf*)(brow + kt * 32 + hi * 16);
    acc = wmma_bf16(cat8(a0, a1), b, acc);
  }
  const float bv = fcb[lo];
#pragma unroll
  for (int vv = 0; vv < 8; ++vv) {
    const int mm = mt * 16 + vv + 8 * hi;
    feats[(size_t)mm * K_TAG + lo] = acc[vv] + bv;
  }
}

// ---------------- phase 6: Viterbi (1 wave, backpointers in LDS) ----------------
__global__ __launch_bounds__(32) void k_viterbi(const float* __restrict__ feats,
                                                const float* __restrict__ trans,
                                                float* __restrict__ out) {
  extern __shared__ char smem[];
  int* bps = (int*)smem;                    // [T][16] = 256 KB
  const int lane = threadIdx.x;
  const int ln = lane & 15;
  float trow[K_TAG];
#pragma unroll
  for (int p = 0; p < K_TAG; ++p) trow[p] = trans[ln * K_TAG + p];
  float v = (lane == START_TAG) ? 0.0f : NEG_INF;

  for (int t = 0; t < T_LEN; ++t) {
    float feat = (lane < K_TAG) ? feats[t * K_TAG + lane] : 0.0f;
    float best = -3.4e38f;
    int bi = 0;
#pragma unroll
    for (int p = 0; p < K_TAG; ++p) {
      float vp = __shfl(v, p, 32);
      float sc = vp + trow[p];
      if (sc > best) { best = sc; bi = p; }
    }
    if (lane < K_TAG) {
      bps[t * K_TAG + lane] = bi;
      v = best + feat;
    }
  }
  float term = (lane < K_TAG) ? (v + trans[END_TAG * K_TAG + lane]) : -3.4e38f;
  float bestt = -3.4e38f;
  int bi = 0;
#pragma unroll
  for (int p = 0; p < K_TAG; ++p) {
    float tp = __shfl(term, p, 32);
    if (lane == 0 && tp > bestt) { bestt = tp; bi = p; }
  }
  if (lane == 0) {
    out[0] = bestt;
    int tag = bi;
    for (int t = T_LEN - 1; t >= 0; --t) {
      out[1 + t] = (float)tag;
      tag = bps[t * K_TAG + tag];
    }
  }
}

// ---------------- host launch ----------------
extern "C" void kernel_launch(void* const* d_in, const int* in_sizes, int n_in,
                              void* d_out, int out_size, void* d_ws, size_t ws_size,
                              hipStream_t stream) {
  (void)in_sizes; (void)n_in; (void)out_size; (void)ws_size;
  const int*   x     = (const int*)  d_in[0];
  const float* h0    = (const float*)d_in[1];
  const float* c0    = (const float*)d_in[2];
  const float* embed = (const float*)d_in[3];
  const float* Wih_f = (const float*)d_in[4];
  const float* Whh_f = (const float*)d_in[5];
  const float* b_f   = (const float*)d_in[6];
  const float* Wih_b = (const float*)d_in[7];
  const float* Whh_b = (const float*)d_in[8];
  const float* b_b   = (const float*)d_in[9];
  const float* fc_w  = (const float*)d_in[10];
  const float* fc_b  = (const float*)d_in[11];
  const float* trans = (const float*)d_in[12];
  float* out = (float*)d_out;

  char* ws = (char*)d_ws;
  size_t off = 0;
  auto take = [&](size_t bytes) -> char* {
    char* p = ws + off;
    off += (bytes + 255) & ~(size_t)255;
    return p;
  };

  __bf16* wihbf_f = (__bf16*)take((size_t)G4 * E_DIM * 2);
  __bf16* wihbf_b = (__bf16*)take((size_t)G4 * E_DIM * 2);
  __bf16* whhbf_f = (__bf16*)take((size_t)G4 * H_DIM * 2);
  __bf16* whhbf_b = (__bf16*)take((size_t)G4 * H_DIM * 2);
  __bf16* fcwbf   = (__bf16*)take((size_t)K_TAG * 2 * H_DIM * 2);
  __bf16* embbf   = (__bf16*)take((size_t)T_LEN * E_DIM * 2);
  float*  Gf      = (float*)take((size_t)T_LEN * G4 * 4);
  float*  Gb      = (float*)take((size_t)T_LEN * G4 * 4);
  __bf16* hbf_f   = (__bf16*)take((size_t)(T_LEN + 1) * H_DIM * 2);
  __bf16* hbf_b   = (__bf16*)take((size_t)(T_LEN + 1) * H_DIM * 2);
  float*  feats   = (float*)take((size_t)T_LEN * K_TAG * 4);
  int*    ready   = (int*)take((size_t)2 * (T_LEN + 1) * RDY_STRIDE * 4);

  const int recLds = 131072 + 1024 + 512;      // Whh slice + h_bf16 + y
  const int vitLds = T_LEN * K_TAG * 4;        // backpointers
  hipFuncSetAttribute((const void*)k_rec, hipFuncAttributeMaxDynamicSharedMemorySize, recLds);
  hipFuncSetAttribute((const void*)k_viterbi, hipFuncAttributeMaxDynamicSharedMemorySize, vitLds);

  k_cvt_rm<<<2048, 256, 0, stream>>>(wihbf_f, Wih_f, G4 * E_DIM);
  k_cvt_rm<<<2048, 256, 0, stream>>>(wihbf_b, Wih_b, G4 * E_DIM);
  k_cvt_rm<<<64, 256, 0, stream>>>(fcwbf, fc_w, K_TAG * 2 * H_DIM);
  k_cvt_whh<<<4096, 256, 0, stream>>>(whhbf_f, Whh_f);
  k_cvt_whh<<<4096, 256, 0, stream>>>(whhbf_b, Whh_b);
  k_gather<<<4096, 256, 0, stream>>>(embbf, embed, x);
  k_gemm_in<<<8192, 256, 0, stream>>>(embbf, wihbf_f, wihbf_b, b_f, b_b, Gf, Gb);
  k_init<<<513, 256, 0, stream>>>(ready, hbf_f, hbf_b, h0);
  k_rec<<<32, 256, recLds, stream>>>(whhbf_f, whhbf_b, Gf, Gb, c0, hbf_f, hbf_b, ready);
  k_gemm_fc<<<32, 256, 0, stream>>>(hbf_f, hbf_b, fcwbf, fc_b, feats);
  k_viterbi<<<1, 32, vitLds, stream>>>(feats, trans, out);
}
